// LongformerEmbed_70531952935559
// MI455X (gfx1250) — matrix operand order repaired and checked by reading, not compiled, not run on hardware
//
#include <hip/hip_runtime.h>
#include <hip/hip_bf16.h>
#include <math.h>

// ---------------- model constants ----------------
#define Sq   2048
#define Hd   768
#define NHn  12
#define HDd  64
#define cW   256
#define NCc  8
#define Ll   12
#define FFf  3072
#define HIDh 256
#define EMBe 256
#define SCALEQ 0.125f
#define EPSL 1e-5f

typedef __attribute__((ext_vector_type(16))) _Float16 v16h;
typedef __attribute__((ext_vector_type(8)))  float    v8f;

// ---------------- WMMA helpers (gfx1250 wave32) ----------------
__device__ __forceinline__ v8f wmma32(v16h a, v16h b, v8f c) {
  // D = A(16x32 f16) * B(32x16 f16) + C(16x16 f32)
  return __builtin_amdgcn_wmma_f32_16x16x32_f16(
      false, a, false, b, (short)0, c, false, false);
}

union frag_cv { uint4 u[2]; v16h v; };
union h8_cv   { uint4 u; _Float16 h[8]; };

// A fragment from row-major LDS tile (ld halfs, 16B-aligned rows).
// lane<16: K kb+{0..7,16..23}; lane>=16: K kb+{8..15,24..31}  -> 2x b128
__device__ __forceinline__ v16h ld_frag_a(const _Float16* s, int ld, int row,
                                          int kb, int lane) {
  const _Float16* r = s + (size_t)row * ld + kb + ((lane >> 4) << 3);
  frag_cv cv;
  cv.u[0] = ((const uint4*)r)[0];         // halfs 0..7
  cv.u[1] = ((const uint4*)(r + 16))[0];  // halfs 16..23
  return cv.v;
}

// B fragment from TRANSPOSED LDS tile s[n][k] (ld halfs).
// lane holds col cb+(lane&15); lane<16: K kb+{0..7,16..23};
// lane>=16: K kb+{8..15,24..31} -> 2x b128
__device__ __forceinline__ v16h ld_frag_bt(const _Float16* s, int ld, int cb,
                                           int kb, int lane) {
  const _Float16* r =
      s + (size_t)(cb + (lane & 15)) * ld + kb + ((lane >> 4) << 3);
  frag_cv cv;
  cv.u[0] = ((const uint4*)r)[0];
  cv.u[1] = ((const uint4*)(r + 16))[0];
  return cv.v;
}

// ---------------- embeddings + LN ----------------
__global__ __launch_bounds__(256) void embed_ln_kernel(
    const int* __restrict__ ids, const float* __restrict__ wemb,
    const float* __restrict__ pemb, const float* __restrict__ gs,
    const float* __restrict__ gb, float* __restrict__ outf,
    _Float16* __restrict__ outh) {
  const int row = blockIdx.x, t = threadIdx.x;
  __shared__ float xs[Hd];
  __shared__ float red[256];
  const size_t wb = (size_t)ids[row] * Hd;
  float loc = 0.f;
  for (int i = t; i < Hd; i += 256) {
    float x = wemb[wb + i] + pemb[(size_t)row * Hd + i];
    xs[i] = x; loc += x;
  }
  red[t] = loc; __syncthreads();
  for (int off = 128; off > 0; off >>= 1) {
    if (t < off) red[t] += red[t + off];
    __syncthreads();
  }
  const float mu = red[0] * (1.f / Hd);
  __syncthreads();
  float lv = 0.f;
  for (int i = t; i < Hd; i += 256) { float d = xs[i] - mu; lv += d * d; }
  red[t] = lv; __syncthreads();
  for (int off = 128; off > 0; off >>= 1) {
    if (t < off) red[t] += red[t + off];
    __syncthreads();
  }
  const float rstd = rsqrtf(red[0] * (1.f / Hd) + EPSL);
  for (int i = t; i < Hd; i += 256) {
    float y = (xs[i] - mu) * rstd * gs[i] + gb[i];
    outf[(size_t)row * Hd + i] = y;
    outh[(size_t)row * Hd + i] = (_Float16)y;
  }
}

// ---------------- residual + LN ----------------
__global__ __launch_bounds__(256) void add_ln_kernel(
    const float* __restrict__ A, const float* __restrict__ Bv,
    const float* __restrict__ gs, const float* __restrict__ gb,
    float* __restrict__ outf, _Float16* __restrict__ outh) {
  const int row = blockIdx.x, t = threadIdx.x;
  __shared__ float xs[Hd];
  __shared__ float red[256];
  float loc = 0.f;
  for (int i = t; i < Hd; i += 256) {
    float x = A[(size_t)row * Hd + i] + Bv[(size_t)row * Hd + i];
    xs[i] = x; loc += x;
  }
  red[t] = loc; __syncthreads();
  for (int off = 128; off > 0; off >>= 1) {
    if (t < off) red[t] += red[t + off];
    __syncthreads();
  }
  const float mu = red[0] * (1.f / Hd);
  __syncthreads();
  float lv = 0.f;
  for (int i = t; i < Hd; i += 256) { float d = xs[i] - mu; lv += d * d; }
  red[t] = lv; __syncthreads();
  for (int off = 128; off > 0; off >>= 1) {
    if (t < off) red[t] += red[t + off];
    __syncthreads();
  }
  const float rstd = rsqrtf(red[0] * (1.f / Hd) + EPSL);
  for (int i = t; i < Hd; i += 256) {
    float y = (xs[i] - mu) * rstd * gs[i] + gb[i];
    if (outf) outf[(size_t)row * Hd + i] = y;
    outh[(size_t)row * Hd + i] = (_Float16)y;
  }
}

// ---------------- WMMA GEMM: Y = act((Xf16 @ Wf32 + b) * scale) ----------
// block: 256 thr = 8 waves, tile 128(M) x 64(N), K step 32. W staged
// transposed (f16) so B fragments are contiguous ds_load_b128 pairs.
__global__ __launch_bounds__(256) void gemm_f16_kernel(
    const _Float16* __restrict__ X, const float* __restrict__ Wt,
    const float* __restrict__ bias, float* __restrict__ Yf,
    _Float16* __restrict__ Yh, int M, int N, int K, int act, float scale) {
  __shared__ _Float16 sX[128][40];
  __shared__ _Float16 sWt[64][40];  // transposed: [n][k]
  const int tid = threadIdx.x, lane = tid & 31, wave = tid >> 5;
  const int bm = blockIdx.y * 128, bn = blockIdx.x * 64;
  v8f zero = {};
  v8f acc[4] = {zero, zero, zero, zero};
  for (int k0 = 0; k0 < K; k0 += 32) {
    {  // stage X tile 128x32 (f16), 2x uint4 per thread
      const int row = tid >> 1, ch = (tid & 1) * 16;
      const uint4* src = (const uint4*)(X + (size_t)(bm + row) * K + k0 + ch);
      uint4* dst = (uint4*)&sX[row][ch];
      dst[0] = src[0]; dst[1] = src[1];
    }
    {  // stage W tile 32x64 fp32 -> f16, transposed into sWt[n][k]
      const int flat = tid * 8, row = flat >> 6, col = flat & 63;  // row = k
      const float* src = Wt + (size_t)(k0 + row) * N + bn + col;
      float4 u0 = ((const float4*)src)[0];
      float4 u1 = ((const float4*)src)[1];
      sWt[col + 0][row] = (_Float16)u0.x;
      sWt[col + 1][row] = (_Float16)u0.y;
      sWt[col + 2][row] = (_Float16)u0.z;
      sWt[col + 3][row] = (_Float16)u0.w;
      sWt[col + 4][row] = (_Float16)u1.x;
      sWt[col + 5][row] = (_Float16)u1.y;
      sWt[col + 6][row] = (_Float16)u1.z;
      sWt[col + 7][row] = (_Float16)u1.w;
    }
    if (k0 + 32 < K)  // stream next weight tile (global_prefetch_b8)
      __builtin_prefetch(Wt + (size_t)(k0 + 32) * N + bn + (tid & 63) * 8, 0, 1);
    __syncthreads();
    v16h a = ld_frag_a(&sX[0][0], 40, wave * 16 + (lane & 15), 0, lane);
#pragma unroll
    for (int j = 0; j < 4; ++j) {
      v16h b = ld_frag_bt(&sWt[0][0], 40, j * 16, 0, lane);
      acc[j] = wmma32(a, b, acc[j]);
    }
    __syncthreads();
  }
  const int mrow = bm + wave * 16 + ((lane >> 4) << 3);
#pragma unroll
  for (int j = 0; j < 4; ++j) {
    const int col = bn + j * 16 + (lane & 15);
    const float bv = bias[col];
#pragma unroll
    for (int r = 0; r < 8; ++r) {
      float y = (acc[j][r] + bv) * scale;
      if (act == 1) y = 0.5f * y * (1.f + erff(y * 0.70710678118f));  // gelu
      else if (act == 2) y = tanhf(y);
      const size_t o = (size_t)(mrow + r) * N + col;
      if (Yf) Yf[o] = y;
      if (Yh) Yh[o] = (_Float16)y;
    }
  }
}

// ---------------- banded attention scores (WMMA Q*K^T) ----------------
// grid (NC, NH). sK[key][d] is already B^T layout for Q*K^T.
__global__ __launch_bounds__(256) void attn_scores_kernel(
    const _Float16* __restrict__ Q, const _Float16* __restrict__ Kh,
    const int* __restrict__ amask, const int* __restrict__ gmask,
    float* __restrict__ Sband, float* __restrict__ Sglob) {
  __shared__ _Float16 sQ[256][72];
  __shared__ _Float16 sK[64][72];
  const int c = blockIdx.x, n = blockIdx.y;
  const int tid = threadIdx.x, lane = tid & 31, wave = tid >> 5;
  {  // stage this thread's own query row (256 rows, 64 dims)
    const uint4* src = (const uint4*)(Q + (size_t)(c * cW + tid) * Hd + n * HDd);
#pragma unroll
    for (int p = 0; p < 8; ++p) ((uint4*)&sQ[tid][p * 8])[0] = src[p];
  }
  {  // global column: q_row . k[0]  (self-staged row, no barrier needed)
    float acc = 0.f;
#pragma unroll
    for (int d = 0; d < HDd; ++d)
      acc += (float)sQ[tid][d] * (float)Kh[n * HDd + d];
    Sglob[(size_t)(n * NCc + c) * cW + tid] = (amask[0] != 0) ? acc : -1e9f;
  }
  for (int kt = 0; kt < 12; ++kt) {  // 12 * 64 = 768 band keys
    __syncthreads();
    {  // stage banded K rows (gather with zero-pad), 2 uint4 per thread
#pragma unroll
      for (int h2 = 0; h2 < 2; ++h2) {
        const int idx = tid + h2 * 256;
        const int row = idx >> 3, part = idx & 7;
        const int j = kt * 64 + row;
        const int kpos = c * cW + j - cW;
        uint4 val = {0u, 0u, 0u, 0u};
        if (kpos >= 0 && kpos < Sq)
          val = ((const uint4*)(Kh + (size_t)kpos * Hd + n * HDd + part * 8))[0];
        ((uint4*)&sK[row][part * 8])[0] = val;
      }
    }
    __syncthreads();
#pragma unroll
    for (int mt = 0; mt < 2; ++mt) {
      const int m0 = wave * 32 + mt * 16;
      v8f zero = {};
      v8f acc[4] = {zero, zero, zero, zero};
#pragma unroll
      for (int ks = 0; ks < 2; ++ks) {  // HD=64 = 2 K-steps of 32
        v16h a = ld_frag_a(&sQ[0][0], 72, m0 + (lane & 15), ks * 32, lane);
#pragma unroll
        for (int j = 0; j < 4; ++j) {
          v16h b = ld_frag_bt(&sK[0][0], 72, j * 16, ks * 32, lane);
          acc[j] = wmma32(a, b, acc[j]);
        }
      }
      const int mrow = m0 + ((lane >> 4) << 3);
#pragma unroll
      for (int j = 0; j < 4; ++j) {
        const int jc = kt * 64 + j * 16 + (lane & 15);
#pragma unroll
        for (int r = 0; r < 8; ++r) {
          const int m = mrow + r;
          const int rel = jc - m;
          const int kpos = c * cW + jc - cW;
          const bool ok = (rel >= 0) && (rel <= 2 * cW) && (kpos >= 0) &&
                          (kpos < Sq) && (amask[kpos] != 0) &&
                          (gmask[kpos] == 0);
          Sband[((size_t)(n * NCc + c) * cW + m) * 768 + jc] =
              ok ? acc[j][r] : -1e9f;
        }
      }
    }
  }
}

// ---------------- softmax over 769 cols, one wave per row ----------------
__global__ __launch_bounds__(256) void softmax_kernel(
    const float* __restrict__ Sband, const float* __restrict__ Sglob,
    _Float16* __restrict__ Pband, float* __restrict__ Pglob) {
  const int tid = threadIdx.x, lane = tid & 31, wave = tid >> 5;
  const size_t row = (size_t)blockIdx.x * 8 + wave;
  const float* sb = Sband + row * 768;
  const float sg = Sglob[row];
  float mx = sg;
  for (int j = lane; j < 768; j += 32) mx = fmaxf(mx, sb[j]);
#pragma unroll
  for (int off = 16; off > 0; off >>= 1) mx = fmaxf(mx, __shfl_xor(mx, off, 32));
  float sum = (lane == 0) ? __expf(sg - mx) : 0.f;
  for (int j = lane; j < 768; j += 32) sum += __expf(sb[j] - mx);
#pragma unroll
  for (int off = 16; off > 0; off >>= 1) sum += __shfl_xor(sum, off, 32);
  const float inv = 1.f / sum;
  for (int j = lane; j < 768; j += 32)
    Pband[row * 768 + j] = (_Float16)(__expf(sb[j] - mx) * inv);
  if (lane == 0) Pglob[row] = __expf(sg - mx) * inv;
}

// ---------------- P*V (WMMA) + global column value ----------------
__global__ __launch_bounds__(256) void attn_pv_kernel(
    const _Float16* __restrict__ Pband, const float* __restrict__ Pglob,
    const _Float16* __restrict__ Vh, _Float16* __restrict__ Out) {
  __shared__ _Float16 sP[256][40];
  __shared__ _Float16 sVt[64][40];  // transposed: [d][key]
  const int c = blockIdx.x, n = blockIdx.y;
  const int tid = threadIdx.x, lane = tid & 31, wave = tid >> 5;
  v8f zero = {};
  v8f acc[2][4];
#pragma unroll
  for (int mt = 0; mt < 2; ++mt)
#pragma unroll
    for (int j = 0; j < 4; ++j) acc[mt][j] = zero;
  for (int kt = 0; kt < 24; ++kt) {  // 24 * 32 = 768 band keys
    {  // stage probabilities: own row, 32 cols
      const uint4* src = (const uint4*)(Pband +
          ((size_t)(n * NCc + c) * cW + tid) * 768 + kt * 32);
#pragma unroll
      for (int p = 0; p < 4; ++p) ((uint4*)&sP[tid][p * 8])[0] = src[p];
    }
    {  // stage banded V rows (gather, zero-pad), transposed into sVt[d][key]
      const int row = tid >> 3, part = tid & 7;  // row = key 0..31
      const int j = kt * 32 + row;
      const int kpos = c * cW + j - cW;
      h8_cv cv;
      cv.u = uint4{0u, 0u, 0u, 0u};
      if (kpos >= 0 && kpos < Sq)
        cv.u = ((const uint4*)(Vh + (size_t)kpos * Hd + n * HDd + part * 8))[0];
#pragma unroll
      for (int i = 0; i < 8; ++i) sVt[part * 8 + i][row] = cv.h[i];
    }
    __syncthreads();
#pragma unroll
    for (int mt = 0; mt < 2; ++mt) {
      v16h a =
          ld_frag_a(&sP[0][0], 40, wave * 32 + mt * 16 + (lane & 15), 0, lane);
#pragma unroll
      for (int j = 0; j < 4; ++j) {
        v16h b = ld_frag_bt(&sVt[0][0], 40, j * 16, 0, lane);
        acc[mt][j] = wmma32(a, b, acc[mt][j]);
      }
    }
    __syncthreads();
  }
#pragma unroll
  for (int mt = 0; mt < 2; ++mt) {
    const int m0 = wave * 32 + mt * 16 + ((lane >> 4) << 3);
#pragma unroll
    for (int j = 0; j < 4; ++j) {
      const int col = j * 16 + (lane & 15);
      const float v0 = (float)Vh[n * HDd + col];  // local v row 0
#pragma unroll
      for (int r = 0; r < 8; ++r) {
        const int m = m0 + r;
        const float pg = Pglob[(size_t)(n * NCc + c) * cW + m];
        const float y = acc[mt][j][r] + pg * v0;
        Out[(size_t)(c * cW + m) * Hd + n * HDd + col] = (_Float16)y;
      }
    }
  }
}

// ---------------- global token attention (tiny), one block per head ------
__global__ __launch_bounds__(256) void global_attn_kernel(
    const float* __restrict__ h, const float* __restrict__ Wqg,
    const float* __restrict__ bqg, const _Float16* __restrict__ KG,
    const _Float16* __restrict__ VG, const int* __restrict__ amask,
    _Float16* __restrict__ attn) {
  const int n = blockIdx.x, t = threadIdx.x;
  __shared__ float sh[Hd];
  __shared__ float sq[HDd];
  __shared__ float ss[Sq];
  __shared__ float red[256];
  for (int i = t; i < Hd; i += 256) sh[i] = h[i];  // row 0 of h
  __syncthreads();
  if (t < HDd) {
    float a = bqg[n * HDd + t];
    for (int k = 0; k < Hd; ++k) a += sh[k] * Wqg[(size_t)k * Hd + n * HDd + t];
    sq[t] = a * SCALEQ;
  }
  __syncthreads();
  for (int s = t; s < Sq; s += 256) {
    float a = 0.f;
#pragma unroll
    for (int d = 0; d < HDd; ++d)
      a += sq[d] * (float)KG[(size_t)s * Hd + n * HDd + d];
    ss[s] = (amask[s] != 0) ? a : -1e9f;
  }
  __syncthreads();
  float mx = -1e30f;
  for (int s = t; s < Sq; s += 256) mx = fmaxf(mx, ss[s]);
  red[t] = mx; __syncthreads();
  for (int off = 128; off > 0; off >>= 1) {
    if (t < off) red[t] = fmaxf(red[t], red[t + off]);
    __syncthreads();
  }
  mx = red[0]; __syncthreads();
  float sum = 0.f;
  for (int s = t; s < Sq; s += 256) {
    float e = __expf(ss[s] - mx);
    ss[s] = e; sum += e;
  }
  red[t] = sum; __syncthreads();
  for (int off = 128; off > 0; off >>= 1) {
    if (t < off) red[t] += red[t + off];
    __syncthreads();
  }
  const float inv = 1.f / red[0];
  if (t < HDd) {
    float a = 0.f;
    for (int s = 0; s < Sq; ++s)
      a += ss[s] * (float)VG[(size_t)s * Hd + n * HDd + t];
    attn[n * HDd + t] = (_Float16)(a * inv);  // overwrite global-token row 0
  }
}

// ---------------- pooled head: tanh(h0@Wd+bd) @ Wo + bo ----------------
__global__ __launch_bounds__(256) void head_kernel(
    const float* __restrict__ h, const float* __restrict__ dw,
    const float* __restrict__ db, const float* __restrict__ ow,
    const float* __restrict__ ob, float* __restrict__ out) {
  __shared__ float p[Hd];
  __shared__ float y[HIDh];
  const int t = threadIdx.x;
  for (int i = t; i < Hd; i += 256) p[i] = h[i];
  __syncthreads();
  {
    float a = db[t];
    for (int k = 0; k < Hd; ++k) a += p[k] * dw[(size_t)k * HIDh + t];
    y[t] = tanhf(a);
  }
  __syncthreads();
  {
    float a = ob[t];
    for (int k = 0; k < HIDh; ++k) a += y[k] * ow[(size_t)k * EMBe + t];
    out[t] = a;
  }
}

// ---------------- launch ----------------
extern "C" void kernel_launch(void* const* d_in, const int* in_sizes, int n_in,
                              void* d_out, int out_size, void* d_ws,
                              size_t ws_size, hipStream_t stream) {
  (void)in_sizes; (void)n_in; (void)out_size; (void)ws_size;
  const int*   ids   = (const int*)d_in[0];
  const int*   amask = (const int*)d_in[1];
  const int*   gmask = (const int*)d_in[2];
  const float* wemb  = (const float*)d_in[3];
  const float* pemb  = (const float*)d_in[4];
  const float* elns  = (const float*)d_in[5];
  const float* elnb  = (const float*)d_in[6];
  const float* Wq  = (const float*)d_in[7];
  const float* bq  = (const float*)d_in[8];
  const float* Wk  = (const float*)d_in[9];
  const float* bk  = (const float*)d_in[10];
  const float* Wv  = (const float*)d_in[11];
  const float* bv  = (const float*)d_in[12];
  const float* Wqg = (const float*)d_in[13];
  const float* bqg = (const float*)d_in[14];
  const float* Wkg = (const float*)d_in[15];
  const float* bkg = (const float*)d_in[16];
  const float* Wvg = (const float*)d_in[17];
  const float* bvg = (const float*)d_in[18];
  const float* Wo  = (const float*)d_in[19];
  const float* bo  = (const float*)d_in[20];
  const float* l1s = (const float*)d_in[21];
  const float* l1b = (const float*)d_in[22];
  const float* Wf1 = (const float*)d_in[23];
  const float* bf1 = (const float*)d_in[24];
  const float* Wf2 = (const float*)d_in[25];
  const float* bf2 = (const float*)d_in[26];
  const float* l2s = (const float*)d_in[27];
  const float* l2b = (const float*)d_in[28];
  const float* dnw = (const float*)d_in[29];
  const float* dnb = (const float*)d_in[30];
  const float* otw = (const float*)d_in[31];
  const float* otb = (const float*)d_in[32];

  char* p = (char*)d_ws;
  auto alloc = [&](size_t n) -> char* {
    char* r = p;
    p += (n + 255) & ~(size_t)255;
    return r;
  };
  const size_t SH = (size_t)Sq * Hd;
  float*    h_f    = (float*)alloc(SH * 4);
  _Float16* h_h    = (_Float16*)alloc(SH * 2);
  float*    h2_f   = (float*)alloc(SH * 4);
  _Float16* h2_h   = (_Float16*)alloc(SH * 2);
  _Float16* q_h    = (_Float16*)alloc(SH * 2);
  _Float16* k_h    = (_Float16*)alloc(SH * 2);
  _Float16* v_h    = (_Float16*)alloc(SH * 2);
  _Float16* kg_h   = (_Float16*)alloc(SH * 2);
  _Float16* vg_h   = (_Float16*)alloc(SH * 2);
  _Float16* attn_h = (_Float16*)alloc(SH * 2);
  float*    proj_f = (float*)alloc(SH * 4);
  _Float16* ff1_h  = (_Float16*)alloc((size_t)Sq * FFf * 2);
  float*    ff2_f  = (float*)alloc(SH * 4);
  const size_t NR = (size_t)NHn * NCc * cW;  // 24576 rows
  float*    Sband  = (float*)alloc(NR * 768 * 4);
  float*    Sglob  = (float*)alloc(NR * 4);
  _Float16* Pband  = (_Float16*)alloc(NR * 768 * 2);
  float*    Pglob  = (float*)alloc(NR * 4);

  embed_ln_kernel<<<Sq, 256, 0, stream>>>(ids, wemb, pemb, elns, elnb, h_f, h_h);

  const dim3 gH(Hd / 64, Sq / 128);    // 768-col GEMMs
  const dim3 gF(FFf / 64, Sq / 128);   // 3072-col GEMM
  const dim3 gAtt(NCc, NHn);

  for (int l = 0; l < Ll; ++l) {
    const size_t oHH = (size_t)l * Hd * Hd, oH = (size_t)l * Hd;
    const size_t oHF = (size_t)l * Hd * FFf, oF = (size_t)l * FFf;
    gemm_f16_kernel<<<gH, 256, 0, stream>>>(h_h, Wq + oHH, bq + oH, nullptr,
                                            q_h, Sq, Hd, Hd, 0, SCALEQ);
    gemm_f16_kernel<<<gH, 256, 0, stream>>>(h_h, Wk + oHH, bk + oH, nullptr,
                                            k_h, Sq, Hd, Hd, 0, 1.f);
    gemm_f16_kernel<<<gH, 256, 0, stream>>>(h_h, Wv + oHH, bv + oH, nullptr,
                                            v_h, Sq, Hd, Hd, 0, 1.f);
    gemm_f16_kernel<<<gH, 256, 0, stream>>>(h_h, Wkg + oHH, bkg + oH, nullptr,
                                            kg_h, Sq, Hd, Hd, 0, 1.f);
    gemm_f16_kernel<<<gH, 256, 0, stream>>>(h_h, Wvg + oHH, bvg + oH, nullptr,
                                            vg_h, Sq, Hd, Hd, 0, 1.f);
    attn_scores_kernel<<<gAtt, 256, 0, stream>>>(q_h, k_h, amask, gmask,
                                                 Sband, Sglob);
    softmax_kernel<<<(unsigned)(NR / 8), 256, 0, stream>>>(Sband, Sglob,
                                                           Pband, Pglob);
    attn_pv_kernel<<<gAtt, 256, 0, stream>>>(Pband, Pglob, v_h, attn_h);
    global_attn_kernel<<<NHn, 256, 0, stream>>>(h_f, Wqg + oHH, bqg + oH,
                                                kg_h, vg_h, amask, attn_h);
    gemm_f16_kernel<<<gH, 256, 0, stream>>>(attn_h, Wo + oHH, bo + oH, proj_f,
                                            nullptr, Sq, Hd, Hd, 0, 1.f);
    add_ln_kernel<<<Sq, 256, 0, stream>>>(h_f, proj_f, l1s + oH, l1b + oH,
                                          h2_f, h2_h);
    gemm_f16_kernel<<<gF, 256, 0, stream>>>(h2_h, Wf1 + oHF, bf1 + oF, nullptr,
                                            ff1_h, Sq, FFf, Hd, 1, 1.f);
    gemm_f16_kernel<<<gH, 256, 0, stream>>>(ff1_h, Wf2 + oHF, bf2 + oH, ff2_f,
                                            nullptr, Sq, Hd, FFf, 0, 1.f);
    add_ln_kernel<<<Sq, 256, 0, stream>>>(h2_f, ff2_f, l2s + oH, l2b + oH,
                                          h_f, h_h);
  }
  head_kernel<<<1, 256, 0, stream>>>(h_f, dnw, dnb, otw, otb, (float*)d_out);
}
